// MambaBlock_53652731462315
// MI455X (gfx1250) — matrix operand order repaired and checked by reading, not compiled, not run on hardware
//
#include <hip/hip_runtime.h>
#include <hip/hip_bf16.h>

// ---------------------------------------------------------------------------
// Mamba block forward for gfx1250 (MI455X), wave32 + v_wmma_f32_16x16x32_bf16.
// Dims: H=1024, E=2048, N=16, R=64, B=4, SEQ=2048, M=B*SEQ=8192.
// GEMM K-loop is software-pipelined (register double buffering) so the
// compiler can issue partial s_wait_loadcnt and overlap WMMA with the next
// iteration's global_load_b128 batch instead of draining loadcnt to 0.
// ---------------------------------------------------------------------------

#define MB_H   1024
#define MB_E   2048
#define MB_N2  4096      // 2*E
#define MB_R   64
#define MB_NS  16
#define MB_B   4
#define MB_SEQ 2048
#define MB_M   (MB_B * MB_SEQ)   // 8192 rows

typedef __attribute__((ext_vector_type(16))) __bf16 bfx16;
typedef __attribute__((ext_vector_type(8)))  __bf16 bfx8;
typedef __attribute__((ext_vector_type(8)))  float  fx8;

__device__ __forceinline__ __bf16 f2bf(float f) {
    unsigned int u = __builtin_bit_cast(unsigned int, f);
    unsigned int r = (u + 0x7FFFu + ((u >> 16) & 1u)) >> 16;   // RNE
    unsigned short s = (unsigned short)r;
    return __builtin_bit_cast(__bf16, s);
}
__device__ __forceinline__ float bf2f(__bf16 b) {
    unsigned short s = __builtin_bit_cast(unsigned short, b);
    unsigned int u = ((unsigned int)s) << 16;
    return __builtin_bit_cast(float, u);
}
__device__ __forceinline__ float silu_f(float v) {
    return v / (1.0f + __expf(-v));
}

// ---------------------------------------------------------------------------
// Weight convert+transpose: W[K,N] f32 (row-major) -> Wt[N,K] bf16 (K-major).
// ---------------------------------------------------------------------------
__global__ void mb_w2bf_t_kernel(const float* __restrict__ W, __bf16* __restrict__ Wt,
                                 int K, int N) {
    int idx = blockIdx.x * 256 + threadIdx.x;
    if (idx >= K * N) return;
    int k = idx / N;
    int n = idx - k * N;
    Wt[(size_t)n * K + k] = f2bf(W[idx]);
}

// ---------------------------------------------------------------------------
// LayerNorm over H, output bf16. One block (256 thr) per row.
// ---------------------------------------------------------------------------
__global__ void mb_layernorm_kernel(const float* __restrict__ x,
                                    const float* __restrict__ g,
                                    const float* __restrict__ b,
                                    __bf16* __restrict__ out, int H) {
    __shared__ float red[256];
    int row = blockIdx.x;
    const float* xr = x + (size_t)row * H;
    int tid = threadIdx.x;

    float s = 0.f;
    for (int i = tid; i < H; i += 256) s += xr[i];
    red[tid] = s; __syncthreads();
    for (int off = 128; off > 0; off >>= 1) {
        if (tid < off) red[tid] += red[tid + off];
        __syncthreads();
    }
    float mu = red[0] / (float)H;
    __syncthreads();

    float v = 0.f;
    for (int i = tid; i < H; i += 256) { float d = xr[i] - mu; v += d * d; }
    red[tid] = v; __syncthreads();
    for (int off = 128; off > 0; off >>= 1) {
        if (tid < off) red[tid] += red[tid + off];
        __syncthreads();
    }
    float rstd = rsqrtf(red[0] / (float)H + 1e-5f);

    __bf16* orow = out + (size_t)row * H;
    for (int i = tid; i < H; i += 256)
        orow[i] = f2bf((xr[i] - mu) * rstd * g[i] + b[i]);
}

// ---------------------------------------------------------------------------
// Generic WMMA bf16 GEMM:  D[M,N] = A[M,K] * Bt[N,K]^T  (+ bias, epilogue)
//   A  : bf16 row-major (K contiguous)       -> WMMA A fragment (16x32)
//   Bt : bf16 K-major per output column      -> WMMA B fragment (32x16)
// One wave computes NT adjacent 16x16 tiles sharing the A fragment.
// K-loop is register double-buffered: next iteration's fragment loads are
// issued before the current iteration's WMMAs so waits are partial.
// Epilogue modes:
//   0: v = acc + bias     -> out0(f32, ld=N) and/or out1(bf16, ld=N)
//   1: split-SiLU (N2)    -> n<N/2: silu->out1(bf16,ld=N/2); else silu->out2
//   2: softplus           -> out0(f32)
//   3: v + bias + aux[m,n]-> out0(f32)   (residual add)
// ---------------------------------------------------------------------------
template <int NT>
__global__ void mb_wmma_gemm_kernel(const __bf16* __restrict__ A,
                                    const __bf16* __restrict__ Bt,
                                    const float* __restrict__ bias,
                                    int M, int N, int K, int mode,
                                    float* __restrict__ out0,
                                    __bf16* __restrict__ out1,
                                    __bf16* __restrict__ out2,
                                    const float* __restrict__ aux) {
    const int lane   = threadIdx.x & 31;
    const int waveid = blockIdx.x * 8 + (threadIdx.x >> 5);
    const int tiles_n = N / (16 * NT);
    const int tile_m = waveid / tiles_n;
    const int tile_n = waveid - tile_m * tiles_n;
    if (tile_m >= M / 16) return;

    const int m0 = tile_m * 16;
    const int n0 = tile_n * 16 * NT;
    const int col   = lane & 15;   // A-row within tile / B-column / D-column
    const int rhalf = lane >> 4;   // lane-half selector per ISA layout

    const __bf16* Arow = A + (size_t)(m0 + col) * K;
    const __bf16* Brow[NT];
#pragma unroll
    for (int nt = 0; nt < NT; ++nt)
        Brow[nt] = Bt + (size_t)(n0 + nt * 16 + col) * K + rhalf * 16;

    // Fragment loaders. A (contiguous k order, not the ISA table order): lanes 0-15 hold K[0..15],
    // lanes 16-31 hold K[16..31], row m = lane&15.
    // B, same k order as A: lane holds column n = lane&15, K=[0..15] (lanes 0-15)
    // or K=[16..31] (lanes 16-31) -> contiguous in K-major Bt.
    auto loadA = [&](int k) {
        bfx8 lo = *(const bfx8*)(Arow + k + rhalf * 16);
        bfx8 hi = *(const bfx8*)(Arow + k + rhalf * 16 + 8);
        bfx16 f;
#pragma unroll
        for (int i = 0; i < 8; ++i) { f[i] = lo[i]; f[i + 8] = hi[i]; }
        return f;
    };
    auto loadB = [&](int nt, int k) {
        bfx8 lo = *(const bfx8*)(Brow[nt] + k);
        bfx8 hi = *(const bfx8*)(Brow[nt] + k + 8);
        bfx16 f;
#pragma unroll
        for (int i = 0; i < 8; ++i) { f[i] = lo[i]; f[i + 8] = hi[i]; }
        return f;
    };

    fx8 acc[NT];
#pragma unroll
    for (int nt = 0; nt < NT; ++nt)
#pragma unroll
        for (int i = 0; i < 8; ++i) acc[nt][i] = 0.f;

    // ---- software pipeline: prologue ----
    bfx16 aCur = loadA(0);
    bfx16 bCur[NT];
#pragma unroll
    for (int nt = 0; nt < NT; ++nt) bCur[nt] = loadB(nt, 0);

    // ---- steady state: issue k+32 loads, then consume k fragments ----
    for (int k = 0; k + 32 < K; k += 32) {
        if (k + 96 < K) __builtin_prefetch(Arow + k + 96, 0, 1);  // global_prefetch_b8
        bfx16 aNxt = loadA(k + 32);
        bfx16 bNxt[NT];
#pragma unroll
        for (int nt = 0; nt < NT; ++nt) bNxt[nt] = loadB(nt, k + 32);

#pragma unroll
        for (int nt = 0; nt < NT; ++nt)
            acc[nt] = __builtin_amdgcn_wmma_f32_16x16x32_bf16(
                false, aCur, false, bCur[nt], (short)0, acc[nt], false, false);

        aCur = aNxt;
#pragma unroll
        for (int nt = 0; nt < NT; ++nt) bCur[nt] = bNxt[nt];
    }

    // ---- epilogue of pipeline: last K-step ----
#pragma unroll
    for (int nt = 0; nt < NT; ++nt)
        acc[nt] = __builtin_amdgcn_wmma_f32_16x16x32_bf16(
            false, aCur, false, bCur[nt], (short)0, acc[nt], false, false);

    // Output: D element (vgpr i, lane) = D[m0 + rhalf*8 + i][n0 + nt*16 + col]
    const int half = N >> 1;
#pragma unroll
    for (int nt = 0; nt < NT; ++nt) {
        const int n = n0 + nt * 16 + col;
        const float bs = bias ? bias[n] : 0.f;
#pragma unroll
        for (int i = 0; i < 8; ++i) {
            const int m = m0 + rhalf * 8 + i;
            float v = acc[nt][i] + bs;
            if (mode == 0) {
                if (out0) out0[(size_t)m * N + n] = v;
                if (out1) out1[(size_t)m * N + n] = f2bf(v);
            } else if (mode == 1) {
                float sv = silu_f(v);
                if (n < half) out1[(size_t)m * half + n] = f2bf(sv);
                else          out2[(size_t)m * half + (n - half)] = f2bf(sv);
            } else if (mode == 2) {
                float sp = (v > 20.f) ? v : log1pf(__expf(v));
                out0[(size_t)m * N + n] = sp;
            } else { // mode 3: bias + residual
                out0[(size_t)m * N + n] = v + aux[(size_t)m * N + n];
            }
        }
    }
}

// ---------------------------------------------------------------------------
// Selective scan. One thread per (b, e) channel; N=16 states in registers.
// Sequential over SEQ; B_t/C_t staged through LDS in chunks of 16 steps.
// Fuses y = (scan + u*D) * silu(gate) and emits bf16 for the output GEMM.
// ---------------------------------------------------------------------------
__global__ void mb_scan_kernel(const float* __restrict__ delta,
                               const __bf16* __restrict__ u_bf,
                               const __bf16* __restrict__ sg_bf,
                               const float* __restrict__ Bm,
                               const float* __restrict__ Cm,
                               const float* __restrict__ A_log,
                               const float* __restrict__ Dv,
                               __bf16* __restrict__ y_bf) {
    const int EB = MB_E / 256;                 // e-blocks per batch
    const int b  = blockIdx.x / EB;
    const int e  = (blockIdx.x % EB) * 256 + threadIdx.x;

    float a[MB_NS], h[MB_NS];
#pragma unroll
    for (int n = 0; n < MB_NS; ++n) {
        a[n] = -__expf(A_log[(size_t)e * MB_NS + n]);  // A = -exp(A_log)
        h[n] = 0.f;
    }
    const float De = Dv[e];

    const int CH = 16;
    __shared__ float bc[CH * 32];              // [t][0..15]=B, [t][16..31]=C

    for (int t0 = 0; t0 < MB_SEQ; t0 += CH) {
        __syncthreads();
        for (int j = threadIdx.x; j < CH * 32; j += 256) {
            int tt = j >> 5, n = j & 31;
            size_t r = (size_t)(b * MB_SEQ + t0 + tt) * MB_NS;
            bc[j] = (n < 16) ? Bm[r + n] : Cm[r + (n - 16)];
        }
        __syncthreads();

#pragma unroll 4
        for (int i = 0; i < CH; ++i) {
            const size_t idx = (size_t)(b * MB_SEQ + t0 + i) * MB_E + e;
            const float d  = delta[idx];
            const float uu = bf2f(u_bf[idx]);
            const float du = d * uu;
            const float* bcr = &bc[i * 32];
            float y = 0.f;
#pragma unroll
            for (int n = 0; n < MB_NS; ++n) {
                float dA = __expf(d * a[n]);
                h[n] = dA * h[n] + du * bcr[n];
                y += bcr[16 + n] * h[n];
            }
            y = (y + uu * De) * bf2f(sg_bf[idx]);
            y_bf[idx] = f2bf(y);
        }
    }
}

// ---------------------------------------------------------------------------
// Host-side launcher.
// ---------------------------------------------------------------------------
static inline size_t mb_align(size_t x) { return (x + 255) & ~(size_t)255; }

extern "C" void kernel_launch(void* const* d_in, const int* in_sizes, int n_in,
                              void* d_out, int out_size, void* d_ws, size_t ws_size,
                              hipStream_t stream) {
    const float* x       = (const float*)d_in[0];
    const float* ln_g    = (const float*)d_in[1];
    const float* ln_b    = (const float*)d_in[2];
    const float* W_in    = (const float*)d_in[3];
    const float* b_in    = (const float*)d_in[4];
    const float* W_delta = (const float*)d_in[5];
    const float* b_delta = (const float*)d_in[6];
    const float* W_dt    = (const float*)d_in[7];
    const float* b_dt    = (const float*)d_in[8];
    const float* W_B     = (const float*)d_in[9];
    const float* b_B     = (const float*)d_in[10];
    const float* W_C     = (const float*)d_in[11];
    const float* b_C     = (const float*)d_in[12];
    const float* A_log   = (const float*)d_in[13];
    const float* Dv      = (const float*)d_in[14];
    const float* W_out   = (const float*)d_in[15];
    const float* b_out   = (const float*)d_in[16];

    // ---- workspace carve-up (~200 MB total) ----
    char* ws = (char*)d_ws;
    size_t off = 0;
    auto carve = [&](size_t bytes) { char* p = ws + off; off = mb_align(off + bytes); return p; };

    __bf16* xn_bf   = (__bf16*)carve((size_t)MB_M * MB_H * 2);
    __bf16* Wt_in   = (__bf16*)carve((size_t)MB_N2 * MB_H * 2);
    __bf16* Wt_dl   = (__bf16*)carve((size_t)MB_R * MB_E * 2);
    __bf16* Wt_dt   = (__bf16*)carve((size_t)MB_E * MB_R * 2);
    __bf16* Wt_B    = (__bf16*)carve((size_t)MB_NS * MB_E * 2);
    __bf16* Wt_C    = (__bf16*)carve((size_t)MB_NS * MB_E * 2);
    __bf16* Wt_out  = (__bf16*)carve((size_t)MB_H * MB_E * 2);
    __bf16* u_bf    = (__bf16*)carve((size_t)MB_M * MB_E * 2);
    __bf16* sg_bf   = (__bf16*)carve((size_t)MB_M * MB_E * 2);
    __bf16* dtlo_bf = (__bf16*)carve((size_t)MB_M * MB_R * 2);
    float*  delta_f = (float*) carve((size_t)MB_M * MB_E * 4);
    float*  Bm_f    = (float*) carve((size_t)MB_M * MB_NS * 4);
    float*  Cm_f    = (float*) carve((size_t)MB_M * MB_NS * 4);
    __bf16* y_bf    = (__bf16*)carve((size_t)MB_M * MB_E * 2);
    (void)ws_size; (void)in_sizes; (void)n_in; (void)out_size;

    dim3 blk(256);
    auto g1 = [](int n) { return dim3((n + 255) / 256); };

    // 1) weights -> bf16, K-major
    mb_w2bf_t_kernel<<<g1(MB_H * MB_N2), blk, 0, stream>>>(W_in,    Wt_in,  MB_H, MB_N2);
    mb_w2bf_t_kernel<<<g1(MB_E * MB_R ), blk, 0, stream>>>(W_delta, Wt_dl,  MB_E, MB_R);
    mb_w2bf_t_kernel<<<g1(MB_R * MB_E ), blk, 0, stream>>>(W_dt,    Wt_dt,  MB_R, MB_E);
    mb_w2bf_t_kernel<<<g1(MB_E * MB_NS), blk, 0, stream>>>(W_B,     Wt_B,   MB_E, MB_NS);
    mb_w2bf_t_kernel<<<g1(MB_E * MB_NS), blk, 0, stream>>>(W_C,     Wt_C,   MB_E, MB_NS);
    mb_w2bf_t_kernel<<<g1(MB_E * MB_H ), blk, 0, stream>>>(W_out,   Wt_out, MB_E, MB_H);

    // 2) layernorm -> xn bf16
    mb_layernorm_kernel<<<MB_M, blk, 0, stream>>>(x, ln_g, ln_b, xn_bf, MB_H);

    // 3) xn @ W_in -> split SiLU -> u_bf, sg_bf        (mode 1, NT=4)
    {
        int waves = (MB_M / 16) * (MB_N2 / 64);
        mb_wmma_gemm_kernel<4><<<waves / 8, blk, 0, stream>>>(
            xn_bf, Wt_in, b_in, MB_M, MB_N2, MB_H, 1, nullptr, u_bf, sg_bf, nullptr);
    }
    // 4) u @ W_delta + b_delta -> dtlow bf16           (mode 0, NT=4)
    {
        int waves = (MB_M / 16) * (MB_R / 64);
        mb_wmma_gemm_kernel<4><<<waves / 8, blk, 0, stream>>>(
            u_bf, Wt_dl, b_delta, MB_M, MB_R, MB_E, 0, nullptr, dtlo_bf, nullptr, nullptr);
    }
    // 5) dtlow @ W_dt + b_dt -> softplus -> delta f32  (mode 2, NT=4)
    {
        int waves = (MB_M / 16) * (MB_E / 64);
        mb_wmma_gemm_kernel<4><<<waves / 8, blk, 0, stream>>>(
            dtlo_bf, Wt_dt, b_dt, MB_M, MB_E, MB_R, 2, delta_f, nullptr, nullptr, nullptr);
    }
    // 6/7) u @ W_B, u @ W_C -> Bm, Cm f32              (mode 0, NT=1; N=16)
    {
        int waves = (MB_M / 16) * (MB_NS / 16);
        mb_wmma_gemm_kernel<1><<<waves / 8, blk, 0, stream>>>(
            u_bf, Wt_B, b_B, MB_M, MB_NS, MB_E, 0, Bm_f, nullptr, nullptr, nullptr);
        mb_wmma_gemm_kernel<1><<<waves / 8, blk, 0, stream>>>(
            u_bf, Wt_C, b_C, MB_M, MB_NS, MB_E, 0, Cm_f, nullptr, nullptr, nullptr);
    }
    // 8) selective scan (fuses +u*D and *silu(gate)) -> y bf16
    mb_scan_kernel<<<MB_B * (MB_E / 256), blk, 0, stream>>>(
        delta_f, u_bf, sg_bf, Bm_f, Cm_f, A_log, Dv, y_bf);

    // 9) y @ W_out + b_out + resid -> d_out f32        (mode 3, NT=4)
    {
        int waves = (MB_M / 16) * (MB_H / 64);
        mb_wmma_gemm_kernel<4><<<waves / 8, blk, 0, stream>>>(
            y_bf, Wt_out, b_out, MB_M, MB_H, MB_E, 3, (float*)d_out, nullptr, nullptr, x);
    }
}
